// SetAbstraction_27917287424403
// MI455X (gfx1250) — matrix operand order repaired and checked by reading, not compiled, not run on hardware
//
#include <hip/hip_runtime.h>
#include <hip/hip_bf16.h>
#include <cstdint>
#include <cstddef>

typedef _Float16 half_t;
typedef __attribute__((ext_vector_type(16))) _Float16 v16h;
typedef __attribute__((ext_vector_type(8)))  _Float16 v8h;
typedef __attribute__((ext_vector_type(8)))  float    v8f;

// Load a 16-half WMMA fragment from two 16-byte chunks at p and p+off halves.
static __device__ __forceinline__ v16h ld_frag(const half_t* p, int off) {
    v8h lo = *((const v8h*)p);
    v8h hi = *((const v8h*)(p + off));
    return __builtin_shufflevector(lo, hi,
        0, 1, 2, 3, 4, 5, 6, 7, 8, 9, 10, 11, 12, 13, 14, 15);
}

// ---------------------------------------------------------------------------
// 1. Transpose input [B,9,N] -> pts0 [B,N,9], xyz0 [B,N,3], and l_xyz[0] out.
// ---------------------------------------------------------------------------
__global__ void k_transpose_in(const float* __restrict__ in, float* __restrict__ pts0,
                               float* __restrict__ xyz0, float* __restrict__ out0) {
    int tid = blockIdx.x * blockDim.x + threadIdx.x;
    const int total = 16 * 9 * 4096;
    if (tid >= total) return;
    int n = tid % 4096;
    int c = (tid / 4096) % 9;
    int b = tid / (9 * 4096);
    float v = in[tid];
    pts0[((size_t)(b * 4096 + n)) * 9 + c] = v;
    if (c < 3) {
        xyz0[((size_t)(b * 4096 + n)) * 3 + c] = v;
        out0[((size_t)(b * 3 + c)) * 4096 + n] = v;
    }
}

// ---------------------------------------------------------------------------
// 2. Farthest point sampling. One block per batch. dist[] lives in LDS.
//    Tie-break = lowest index (matches jnp.argmax first-occurrence).
// ---------------------------------------------------------------------------
__global__ void __launch_bounds__(256) k_fps(const float* __restrict__ xyz,
                                             int* __restrict__ fi, int N, int S) {
    int b = blockIdx.x;
    const float* p = xyz + (size_t)b * N * 3;
    __shared__ float dist[4096];
    __shared__ float rv[256];
    __shared__ int   ri[256];
    int t = threadIdx.x;
    for (int j = t; j < N; j += 256) dist[j] = 1e10f;
    __syncthreads();
    int far = 0;
    for (int i = 0; i < S; ++i) {
        if (t == 0) fi[b * S + i] = far;
        float fx = p[far * 3 + 0], fy = p[far * 3 + 1], fz = p[far * 3 + 2];
        float bv = -1.0f; int bi = 0x7fffffff;
        for (int j = t; j < N; j += 256) {
            float dx = p[j * 3 + 0] - fx;
            float dy = p[j * 3 + 1] - fy;
            float dz = p[j * 3 + 2] - fz;
            float d  = dx * dx + dy * dy + dz * dz;
            float dd = fminf(dist[j], d);
            dist[j] = dd;
            if (dd > bv) { bv = dd; bi = j; }
        }
        rv[t] = bv; ri[t] = bi;
        __syncthreads();
        for (int s = 128; s > 0; s >>= 1) {
            if (t < s) {
                float ov = rv[t + s]; int oi = ri[t + s];
                if (ov > rv[t] || (ov == rv[t] && oi < ri[t])) { rv[t] = ov; ri[t] = oi; }
            }
            __syncthreads();
        }
        far = ri[0];
        __syncthreads();
    }
}

// ---------------------------------------------------------------------------
// 3. Gather new_xyz [B,S,3] and write transposed l_xyz output [B,3,S].
// ---------------------------------------------------------------------------
__global__ void k_newxyz(const float* __restrict__ xyz, const int* __restrict__ fi,
                         float* __restrict__ nxyz, float* __restrict__ outx, int N, int S) {
    int tid = blockIdx.x * blockDim.x + threadIdx.x;
    int total = 16 * S * 3;
    if (tid >= total) return;
    int c = tid % 3;
    int s = (tid / 3) % S;
    int b = tid / (3 * S);
    int id = fi[b * S + s];
    float v = xyz[((size_t)(b * N + id)) * 3 + c];
    nxyz[((size_t)(b * S + s)) * 3 + c] = v;
    outx[((size_t)(b * 3 + c)) * S + s] = v;
}

// ---------------------------------------------------------------------------
// 4. Ball query: first K=32 points (ascending index) with d2 <= r2; pad w/ hit0.
// ---------------------------------------------------------------------------
__global__ void k_ballquery(const float* __restrict__ xyz, const float* __restrict__ nxyz,
                            int* __restrict__ gi, int N, int S, float r2) {
    int tid = blockIdx.x * blockDim.x + threadIdx.x;
    int total = 16 * S;
    if (tid >= total) return;
    int b = tid / S;
    const float* p = xyz + (size_t)b * N * 3;
    float cx = nxyz[((size_t)tid) * 3 + 0];
    float cy = nxyz[((size_t)tid) * 3 + 1];
    float cz = nxyz[((size_t)tid) * 3 + 2];
    int base = tid * 32;
    int cnt = 0;
    for (int j = 0; j < N && cnt < 32; ++j) {
        float dx = p[j * 3 + 0] - cx;
        float dy = p[j * 3 + 1] - cy;
        float dz = p[j * 3 + 2] - cz;
        if (dx * dx + dy * dy + dz * dz <= r2) gi[base + cnt++] = j;
    }
    int pad = (cnt > 0) ? gi[base] : 0;
    for (; cnt < 32; ++cnt) gi[base + cnt] = pad;
}

// ---------------------------------------------------------------------------
// 5. Build padded f16 A-matrix rows: [rel_xyz(3) | feats(D) | zeros] -> CinPad.
// ---------------------------------------------------------------------------
__global__ void k_group(const float* __restrict__ xyz, const float* __restrict__ nxyz,
                        const float* __restrict__ feats, const int* __restrict__ gi,
                        half_t* __restrict__ A, int N, int S, int D, int CinPad) {
    int tid = blockIdx.x * blockDim.x + threadIdx.x;
    int total = 16 * S * 32;
    if (tid >= total) return;
    int s = (tid / 32) % S;
    int b = tid / (32 * S);
    int id = gi[tid];
    const float* pt = xyz + ((size_t)(b * N + id)) * 3;
    const float* ct = nxyz + ((size_t)(b * S + s)) * 3;
    half_t* row = A + (size_t)tid * CinPad;
    row[0] = (half_t)(pt[0] - ct[0]);
    row[1] = (half_t)(pt[1] - ct[1]);
    row[2] = (half_t)(pt[2] - ct[2]);
    const float* f = feats + ((size_t)(b * N + id)) * D;
    for (int c = 0; c < D; ++c) row[3 + c] = (half_t)f[c];
    for (int c = 3 + D; c < CinPad; ++c) row[c] = (half_t)0.0f;
}

// ---------------------------------------------------------------------------
// Weight conversion: W [Cout,Cin] f32 -> Wh [Cout,CinPad] f16, zero padded.
// ---------------------------------------------------------------------------
__global__ void k_wconv(const float* __restrict__ W, half_t* __restrict__ Wh,
                        int Cout, int Cin, int CinPad) {
    int tid = blockIdx.x * blockDim.x + threadIdx.x;
    int total = Cout * CinPad;
    if (tid >= total) return;
    int ci = tid % CinPad, co = tid / CinPad;
    Wh[tid] = (ci < Cin) ? (half_t)W[co * Cin + ci] : (half_t)0.0f;
}

// ---------------------------------------------------------------------------
// 6. WMMA GEMM + bias + ReLU, software-pipelined K loop.
//    One wave owns a 16 x (NT*16) output strip; fragments for k-step kk+32 are
//    prefetched before the WMMA group of k-step kk issues, so global loads
//    overlap the matrix pipe instead of serializing on s_wait_loadcnt.
//    A fragment (16x32 f16): lanes 0-15 row M=lr hold K{kk..kk+7, kk+16..kk+23}
//                            lanes 16-31 hold K{kk+8..kk+15, kk+24..kk+31}
//    B fragment (32x16 f16): lane lr = col, same K split as A
//    C/D: VGPR v, lane-half h -> row = 16*mt + 8*h + v, col = n_base + lr.
// ---------------------------------------------------------------------------
template <int NT>
__global__ void __launch_bounds__(256) k_gemm_wmma(
    const half_t* __restrict__ A, const half_t* __restrict__ Wh,
    const float* __restrict__ bias, half_t* __restrict__ out,
    int M, int CinPad, int Cout) {
    int wave = (blockIdx.x * blockDim.x + threadIdx.x) >> 5;
    int lane = threadIdx.x & 31;
    int mt = wave;
    if (mt * 16 >= M) return;                 // wave-uniform exit, EXEC stays full
    int hid = lane >> 4;                      // lane half
    int lr  = lane & 15;
    const half_t* arow = A + ((size_t)(mt * 16 + lr)) * CinPad + hid * 8;

    for (int n0 = 0; n0 < Cout; n0 += NT * 16) {
        const half_t* brow[NT];
#pragma unroll
        for (int nt = 0; nt < NT; ++nt)
            brow[nt] = Wh + (size_t)(n0 + nt * 16 + lr) * CinPad + hid * 8;

        v8f acc[NT];
#pragma unroll
        for (int nt = 0; nt < NT; ++nt) acc[nt] = (v8f){};

        // ---- prologue: fragments for kk = 0 ----
        v16h av = ld_frag(arow, 16);
        v16h bv[NT];
#pragma unroll
        for (int nt = 0; nt < NT; ++nt) bv[nt] = ld_frag(brow[nt], 16);

        // ---- pipelined main loop ----
        for (int kk = 32; kk < CinPad; kk += 32) {
            v16h av_n = ld_frag(arow + kk, 16);
            v16h bv_n[NT];
#pragma unroll
            for (int nt = 0; nt < NT; ++nt) bv_n[nt] = ld_frag(brow[nt] + kk, 16);
#pragma unroll
            for (int nt = 0; nt < NT; ++nt)
                acc[nt] = __builtin_amdgcn_wmma_f32_16x16x32_f16(
                    false, av, false, bv[nt], (short)0, acc[nt], false, false);
            av = av_n;
#pragma unroll
            for (int nt = 0; nt < NT; ++nt) bv[nt] = bv_n[nt];
        }
        // ---- epilogue WMMA group ----
#pragma unroll
        for (int nt = 0; nt < NT; ++nt)
            acc[nt] = __builtin_amdgcn_wmma_f32_16x16x32_f16(
                false, av, false, bv[nt], (short)0, acc[nt], false, false);

        // ---- bias + ReLU + store (f16 feeds the next layer's A matrix) ----
#pragma unroll
        for (int nt = 0; nt < NT; ++nt) {
            int col = n0 + nt * 16 + lr;
            float bs = bias[col];
#pragma unroll
            for (int v = 0; v < 8; ++v) {
                int row = mt * 16 + hid * 8 + v;
                float val = acc[nt][v] + bs;
                val = val > 0.0f ? val : 0.0f;
                out[(size_t)row * Cout + col] = (half_t)val;
            }
        }
    }
}

// ---------------------------------------------------------------------------
// 7. Max over K=32 neighbors -> pooled [B,S,C] f32 and transposed out [B,C,S].
// ---------------------------------------------------------------------------
__global__ void k_maxpool(const half_t* __restrict__ L, float* __restrict__ pooled,
                          float* __restrict__ outp, int S, int C) {
    int tid = blockIdx.x * blockDim.x + threadIdx.x;
    int total = 16 * S * C;
    if (tid >= total) return;
    int c = tid % C;
    int s = (tid / C) % S;
    int b = tid / (C * S);
    const half_t* base = L + ((size_t)(b * S + s) * 32) * C + c;
    float m = (float)base[0];
    for (int k = 1; k < 32; ++k) m = fmaxf(m, (float)base[(size_t)k * C]);
    pooled[((size_t)(b * S + s)) * C + c] = m;
    outp[((size_t)b * C + c) * S + s] = m;
}

// ---------------------------------------------------------------------------
static void launch_gemm(const half_t* A, const half_t* Wh, const float* bias,
                        half_t* out, int M, int CinPad, int Cout, hipStream_t st) {
    int waves = M / 16;
    int blocks = (waves + 7) / 8;
    if (Cout % 64 == 0)
        k_gemm_wmma<4><<<blocks, 256, 0, st>>>(A, Wh, bias, out, M, CinPad, Cout);
    else
        k_gemm_wmma<2><<<blocks, 256, 0, st>>>(A, Wh, bias, out, M, CinPad, Cout);
}

extern "C" void kernel_launch(void* const* d_in, const int* in_sizes, int n_in,
                              void* d_out, int out_size, void* d_ws, size_t ws_size,
                              hipStream_t stream) {
    (void)in_sizes; (void)n_in; (void)out_size; (void)ws_size;
    const float* xyz_in = (const float*)d_in[0];
    float* outf = (float*)d_out;

    // ------- workspace carve-out -------
    char* w = (char*)d_ws;
    auto alloc = [&](size_t bytes) -> void* {
        void* p = (void*)w;
        w += (bytes + 255) & ~(size_t)255;
        return p;
    };
    float* pts0 = (float*)alloc((size_t)16 * 4096 * 9 * 4);
    float* xyz0 = (float*)alloc((size_t)16 * 4096 * 3 * 4);
    const int Slv[4] = {1024, 256, 64, 16};
    float* nxyz[4];
    for (int l = 0; l < 4; ++l) nxyz[l] = (float*)alloc((size_t)16 * Slv[l] * 3 * 4);
    int* fi = (int*)alloc((size_t)16 * 1024 * 4);
    int* gi = (int*)alloc((size_t)16 * 1024 * 32 * 4);
    const int Cclv[4] = {64, 128, 256, 512};
    float* pooled[4];
    for (int l = 0; l < 4; ++l) pooled[l] = (float*)alloc((size_t)16 * Slv[l] * Cclv[l] * 4);
    const int Cin_[12]    = {12, 32, 32, 67, 64, 64, 131, 128, 128, 259, 256, 256};
    const int Cout_[12]   = {32, 32, 64, 64, 64, 128, 128, 128, 256, 256, 256, 512};
    const int CinPadL[12] = {32, 32, 32, 96, 64, 64, 160, 128, 128, 288, 256, 256};
    half_t* wh[12];
    for (int i = 0; i < 12; ++i) wh[i] = (half_t*)alloc((size_t)Cout_[i] * CinPadL[i] * 2);
    half_t* bufA = (half_t*)alloc((size_t)524288 * 32 * 2);   // max(M*CinPad, M*Ca)
    half_t* bufB = (half_t*)alloc((size_t)524288 * 64 * 2);   // max(M*Ca, M*Cc)

    // ------- weight conversion (f32 -> padded f16) -------
    for (int i = 0; i < 12; ++i) {
        int total = Cout_[i] * CinPadL[i];
        k_wconv<<<(total + 255) / 256, 256, 0, stream>>>(
            (const float*)d_in[1 + 2 * i], wh[i], Cout_[i], Cin_[i], CinPadL[i]);
    }

    // ------- input transpose + l_xyz[0] output -------
    {
        int total = 16 * 9 * 4096;
        k_transpose_in<<<(total + 255) / 256, 256, 0, stream>>>(xyz_in, pts0, xyz0, outf);
    }

    // ------- per-level set abstraction -------
    const float Rlv[4]      = {0.1f, 0.2f, 0.4f, 0.8f};
    const int   Dlv[4]      = {9, 64, 128, 256};
    const int   CinPadlv[4] = {32, 96, 160, 288};
    const int   Calv[4]     = {32, 64, 128, 256};
    const size_t oXyz[4] = {196608, 245760, 258048, 261120};
    const size_t oPts[4] = {261888, 1310464, 1834752, 2096896};

    for (int l = 0; l < 4; ++l) {
        int S   = Slv[l];
        int Nin = (l == 0) ? 4096 : Slv[l - 1];
        const float* xin = (l == 0) ? xyz0 : nxyz[l - 1];
        const float* fin = (l == 0) ? pts0 : pooled[l - 1];
        int M = 16 * S * 32;

        k_fps<<<16, 256, 0, stream>>>(xin, fi, Nin, S);

        int t1 = 16 * S * 3;
        k_newxyz<<<(t1 + 255) / 256, 256, 0, stream>>>(xin, fi, nxyz[l], outf + oXyz[l], Nin, S);

        int t2 = 16 * S;
        k_ballquery<<<(t2 + 255) / 256, 256, 0, stream>>>(xin, nxyz[l], gi, Nin, S,
                                                          Rlv[l] * Rlv[l]);

        int t3 = 16 * S * 32;
        k_group<<<(t3 + 255) / 256, 256, 0, stream>>>(xin, nxyz[l], fin, gi, bufA, Nin, S,
                                                      Dlv[l], CinPadlv[l]);

        launch_gemm(bufA, wh[3 * l + 0], (const float*)d_in[2 + 6 * l], bufB,
                    M, CinPadlv[l], Calv[l], stream);
        launch_gemm(bufB, wh[3 * l + 1], (const float*)d_in[4 + 6 * l], bufA,
                    M, Calv[l], Calv[l], stream);
        launch_gemm(bufA, wh[3 * l + 2], (const float*)d_in[6 + 6 * l], bufB,
                    M, Calv[l], Cclv[l], stream);

        int t4 = 16 * S * Cclv[l];
        k_maxpool<<<(t4 + 255) / 256, 256, 0, stream>>>(bufB, pooled[l], outf + oPts[l],
                                                        S, Cclv[l]);
    }
}
